// MultiHeadAttention_46437186404507
// MI455X (gfx1250) — matrix operand order repaired and checked by reading, not compiled, not run on hardware
//
#include <hip/hip_runtime.h>
#include <stdint.h>

// ---------------------------------------------------------------------------
// Types for CDNA5 WMMA (gfx1250, wave32)
// ---------------------------------------------------------------------------
typedef __attribute__((ext_vector_type(16))) __bf16 v16bf;
typedef __attribute__((ext_vector_type(8)))  float  v8f;

union BF16Frag {
    v16bf v;
    uint4 q[2];
};

__device__ __forceinline__ unsigned short f2bf(float f) {
    union { float f; unsigned int u; } x;
    x.f = f;
    unsigned int u = x.u;
    unsigned int r = u + 0x7FFFu + ((u >> 16) & 1u);   // round-to-nearest-even
    return (unsigned short)(r >> 16);
}

__device__ __forceinline__ uint4 ld16(const void* p) {
    return *reinterpret_cast<const uint4*>(p);
}

__device__ __forceinline__ void storeOut(float* p, float v) { *p = v; }
__device__ __forceinline__ void storeOut(unsigned short* p, float v) { *p = f2bf(v); }

__device__ __forceinline__ float rmax16(float v) {
#pragma unroll
    for (int o = 8; o >= 1; o >>= 1) v = fmaxf(v, __shfl_xor(v, o, 32));
    return v;
}
__device__ __forceinline__ float rsum16(float v) {
#pragma unroll
    for (int o = 8; o >= 1; o >>= 1) v += __shfl_xor(v, o, 32);
    return v;
}

// ---------------------------------------------------------------------------
// CDNA5 async global->LDS copy (ASYNCcnt path).
//   LDS address = low 32 bits of the flat pointer (ISA 10.2)
//   Global address = flat 64-bit address (identical in global aperture)
// ---------------------------------------------------------------------------
__device__ __forceinline__ void async_ld16(void* lds_dst, const void* gsrc) {
    unsigned laddr = (unsigned)(uintptr_t)lds_dst;
    unsigned long long gaddr = (unsigned long long)(uintptr_t)gsrc;
    asm volatile("global_load_async_to_lds_b128 %0, %1, off"
                 :: "v"(laddr), "v"(gaddr) : "memory");
}
__device__ __forceinline__ void wait_async0() {
    asm volatile("s_wait_asynccnt 0x0" ::: "memory");
}

// ---------------------------------------------------------------------------
// Kernel 1: fp32 -> bf16 conversion (vectorized, 8 elements / thread)
// ---------------------------------------------------------------------------
__global__ __launch_bounds__(256) void cvt_f32_bf16(const float* __restrict__ src,
                                                    unsigned short* __restrict__ dst,
                                                    int n) {
    int i = (blockIdx.x * 256 + threadIdx.x) * 8;
    if (i + 8 <= n) {
        float4 a = *reinterpret_cast<const float4*>(src + i);
        float4 b = *reinterpret_cast<const float4*>(src + i + 4);
        unsigned short o[8];
        o[0] = f2bf(a.x); o[1] = f2bf(a.y); o[2] = f2bf(a.z); o[3] = f2bf(a.w);
        o[4] = f2bf(b.x); o[5] = f2bf(b.y); o[6] = f2bf(b.z); o[7] = f2bf(b.w);
        *reinterpret_cast<uint4*>(dst + i) = *reinterpret_cast<uint4*>(o);
    }
}

// ---------------------------------------------------------------------------
// Kernel 2/4: C[n,m] = sum_k A[n,k]*W[m,k] + bias[m]
//   A: (Nrows x K) bf16 row-major, W: (Mcols x K) bf16 row-major (torch Linear)
//   Tile: 128x128, BK=32, 8 waves (2x4); async double-buffered LDS staging.
// ---------------------------------------------------------------------------
#define BK       32
#define LDS_STR  40   // 32 + 8 pad (ushorts); rows stay 16B aligned

template <typename OutT>
__global__ __launch_bounds__(256) void gemm_bias(const unsigned short* __restrict__ A,
                                                 const unsigned short* __restrict__ W,
                                                 const float* __restrict__ bias,
                                                 OutT* __restrict__ C,
                                                 int Nrows, int Kdim, int Mcols) {
    __shared__ unsigned short As[2][128 * LDS_STR];
    __shared__ unsigned short Bs[2][128 * LDS_STR];

    const int tid  = threadIdx.x;
    const int lane = tid & 31;
    const int wave = tid >> 5;
    const int hl   = lane >> 4;      // 0: lanes 0-15, 1: lanes 16-31
    const int l16  = lane & 15;
    const int wr   = wave >> 2;      // 0..1 : 64-row group
    const int wc   = wave & 3;       // 0..3 : 32-col group
    const int rowBase = blockIdx.y * 128;
    const int colBase = blockIdx.x * 128;

    // async-stage a 128x32 A tile and 128x32 W tile into buffer bf
    auto stage = [&](int k0, int bf) {
#pragma unroll
        for (int i = 0; i < 2; i++) {
            int c = tid + i * 256;           // 0..511
            int r = c >> 2;
            int cc = (c & 3) * 8;
            async_ld16(&As[bf][r * LDS_STR + cc],
                       A + (size_t)(rowBase + r) * Kdim + k0 + cc);
            async_ld16(&Bs[bf][r * LDS_STR + cc],
                       W + (size_t)(colBase + r) * Kdim + k0 + cc);
        }
    };

    v8f acc[4][2];
    const v8f vzero = {0.f, 0.f, 0.f, 0.f, 0.f, 0.f, 0.f, 0.f};
#pragma unroll
    for (int mt = 0; mt < 4; mt++)
#pragma unroll
        for (int nt = 0; nt < 2; nt++) acc[mt][nt] = vzero;

    // prologue: fill buffer 0
    stage(0, 0);
    wait_async0();
    __syncthreads();

    for (int k0 = 0; k0 < Kdim; k0 += BK) {
        const int cur = (k0 >> 5) & 1;
        if (k0 + BK < Kdim) stage(k0 + BK, cur ^ 1);   // overlap with compute

        // B fragments: lane = column n; lanes<16 K=0..15, lanes>=16 K=16..31
        BF16Frag bf[2];
#pragma unroll
        for (int nt = 0; nt < 2; nt++) {
            int n = wc * 32 + nt * 16 + l16;
            int kc = hl * 16;
            bf[nt].q[0] = ld16(&Bs[cur][n * LDS_STR + kc]);
            bf[nt].q[1] = ld16(&Bs[cur][n * LDS_STR + kc + 8]);
        }
        // A fragments: lane = row m; lanes<16 K=0..15, lanes>=16 K=16..31
#pragma unroll
        for (int mt = 0; mt < 4; mt++) {
            BF16Frag af;
            int r  = wr * 64 + mt * 16 + l16;
            int kc = hl * 16;
            af.q[0] = ld16(&As[cur][r * LDS_STR + kc]);
            af.q[1] = ld16(&As[cur][r * LDS_STR + kc + 8]);
#pragma unroll
            for (int nt = 0; nt < 2; nt++) {
                acc[mt][nt] = __builtin_amdgcn_wmma_f32_16x16x32_bf16(
                    false, af.v, false, bf[nt].v, (short)0, acc[mt][nt], false, false);
            }
        }

        wait_async0();     // own async writes into next buffer complete
        __syncthreads();   // publish to all waves / protect current buffer
    }

    // epilogue: C layout -> row = r + 8*hl, col = l16 within each 16x16 tile
#pragma unroll
    for (int nt = 0; nt < 2; nt++) {
        int col = colBase + wc * 32 + nt * 16 + l16;
        float bv = bias[col];
#pragma unroll
        for (int mt = 0; mt < 4; mt++) {
            int row = rowBase + wr * 64 + mt * 16 + hl * 8;
#pragma unroll
            for (int r = 0; r < 8; r++) {
                float v = acc[mt][nt][r] + bv;
                storeOut(&C[(size_t)(row + r) * Mcols + col], v);
            }
        }
    }
}

// ---------------------------------------------------------------------------
// Kernel 3: flash attention
//   block = (b, h, 128 q-rows); 8 waves, each owns a 16-row q-tile
//   32-key slabs, double-buffered: K via async-to-LDS, V via transpose staging
// ---------------------------------------------------------------------------
#define S_LEN  2048
#define DKH    64
#define KS_STR 72   // 64 + 8 pad ushorts (rows 144B -> 16B aligned)
#define VT_STR 40   // 32 + 8 pad
#define P_STR  40

__global__ __launch_bounds__(256) void attn_flash(const unsigned short* __restrict__ Qb,
                                                  const unsigned short* __restrict__ Kb,
                                                  const unsigned short* __restrict__ Vb,
                                                  const int* __restrict__ mask,
                                                  unsigned short* __restrict__ Ob) {
    __shared__ unsigned short Ks[2][32 * KS_STR];    // 32 keys x 64 dk
    __shared__ unsigned short Vt[2][DKH * VT_STR];   // transposed: [dk][key]
    __shared__ unsigned short Ps[8][16 * P_STR];     // per-wave P staging (16x32)

    const int tid  = threadIdx.x;
    const int lane = tid & 31;
    const int wave = tid >> 5;
    const int hl   = lane >> 4;
    const int l16  = lane & 15;

    const int blk = blockIdx.x;        // 512 = B(2) * H(16) * qgroups(16)
    const int b   = blk >> 8;
    const int h   = (blk >> 4) & 15;
    const int qg  = blk & 15;
    const size_t rowQ = (size_t)b * S_LEN + qg * 128 + wave * 16;

    const int key_ld = tid >> 3;        // 0..31
    const int c8     = (tid & 7) * 8;   // 0..56

    // stage one 32-key slab: K async (verbatim), V through VGPRs (transpose)
    auto stage = [&](int kb2, int bf) {
        size_t grow = ((size_t)b * S_LEN + kb2 + key_ld) * 1024 + h * DKH + c8;
        async_ld16(&Ks[bf][key_ld * KS_STR + c8], Kb + grow);
        uint4 dv = ld16(Vb + grow);
        const unsigned short* pv = reinterpret_cast<const unsigned short*>(&dv);
#pragma unroll
        for (int j = 0; j < 8; j++) Vt[bf][(c8 + j) * VT_STR + key_ld] = pv[j];
    };

    // Q fragments (A layout), resident for the whole key loop
    BF16Frag qf[2];
    {
        const unsigned short* qp = Qb + (rowQ + l16) * 1024 + h * DKH;
        qf[0].q[0] = ld16(qp + hl * 16);
        qf[0].q[1] = ld16(qp + 8 + hl * 16);
        qf[1].q[0] = ld16(qp + 32 + hl * 16);
        qf[1].q[1] = ld16(qp + 40 + hl * 16);
    }

    const v8f vzero = {0.f, 0.f, 0.f, 0.f, 0.f, 0.f, 0.f, 0.f};
    v8f oacc[4];
#pragma unroll
    for (int nt = 0; nt < 4; nt++) oacc[nt] = vzero;
    float m_[8], l_[8];
#pragma unroll
    for (int r = 0; r < 8; r++) { m_[r] = -3.0e38f; l_[r] = 0.f; }

    const float scale = 0.125f;   // 1/sqrt(64)
    const int* mrow = mask + b * S_LEN;

    // prologue: fill buffer 0
    stage(0, 0);
    wait_async0();
    __syncthreads();

    for (int kb = 0; kb < S_LEN; kb += 32) {
        const int cur = (kb >> 5) & 1;
        if (kb + 32 < S_LEN) stage(kb + 32, cur ^ 1);   // overlap with compute

        // ---- scores: two 16x16 tiles (keys kb+0..15, kb+16..31) ----
        v8f s[2];
#pragma unroll
        for (int sub = 0; sub < 2; sub++) {
            BF16Frag kf0, kf1;
            int key = sub * 16 + l16;
            kf0.q[0] = ld16(&Ks[cur][key * KS_STR + hl * 16]);
            kf0.q[1] = ld16(&Ks[cur][key * KS_STR + hl * 16 + 8]);
            kf1.q[0] = ld16(&Ks[cur][key * KS_STR + 32 + hl * 16]);
            kf1.q[1] = ld16(&Ks[cur][key * KS_STR + 32 + hl * 16 + 8]);
            v8f a = vzero;
            a = __builtin_amdgcn_wmma_f32_16x16x32_bf16(false, qf[0].v, false, kf0.v,
                                                        (short)0, a, false, false);
            a = __builtin_amdgcn_wmma_f32_16x16x32_bf16(false, qf[1].v, false, kf1.v,
                                                        (short)0, a, false, false);
            int mv = mrow[kb + sub * 16 + l16];
#pragma unroll
            for (int r = 0; r < 8; r++) {
                float t = a[r] * scale;
                s[sub][r] = mv ? t : -1e9f;
            }
        }

        // ---- online softmax (per row: r + 8*hl; row spans 16 lanes) ----
        float cf[8];
#pragma unroll
        for (int r = 0; r < 8; r++) {
            float mx = rmax16(fmaxf(s[0][r], s[1][r]));
            float mn = fmaxf(m_[r], mx);
            cf[r] = __expf(m_[r] - mn);
            m_[r] = mn;
            float p0 = __expf(s[0][r] - mn);
            float p1 = __expf(s[1][r] - mn);
            s[0][r] = p0; s[1][r] = p1;
            l_[r] = l_[r] * cf[r] + rsum16(p0 + p1);
        }
#pragma unroll
        for (int nt = 0; nt < 4; nt++)
#pragma unroll
            for (int r = 0; r < 8; r++) oacc[nt][r] = oacc[nt][r] * cf[r];

        // ---- P: C-layout regs -> LDS -> A-layout fragment (wave-private) ----
        unsigned short* pp = &Ps[wave][0];
#pragma unroll
        for (int sub = 0; sub < 2; sub++)
#pragma unroll
            for (int r = 0; r < 8; r++)
                pp[(r + hl * 8) * P_STR + sub * 16 + l16] = f2bf(s[sub][r]);
        asm volatile("" ::: "memory");   // keep LDS write->read order (in-order per wave)
        BF16Frag pf;
        pf.q[0] = ld16(&pp[l16 * P_STR + hl * 16]);
        pf.q[1] = ld16(&pp[l16 * P_STR + 8 + hl * 16]);

        // ---- PV: 4 tiles over dk (B frags contiguous from transposed V) ----
#pragma unroll
        for (int nt = 0; nt < 4; nt++) {
            BF16Frag vf;
            int n = nt * 16 + l16;
            vf.q[0] = ld16(&Vt[cur][n * VT_STR + hl * 16]);
            vf.q[1] = ld16(&Vt[cur][n * VT_STR + hl * 16 + 8]);
            oacc[nt] = __builtin_amdgcn_wmma_f32_16x16x32_bf16(
                false, pf.v, false, vf.v, (short)0, oacc[nt], false, false);
        }

        wait_async0();     // own async K writes (next buffer) complete
        __syncthreads();   // publish staging / protect current buffer
    }

    // ---- normalize and store (bf16, combined-heads layout) ----
#pragma unroll
    for (int nt = 0; nt < 4; nt++) {
#pragma unroll
        for (int r = 0; r < 8; r++) {
            float v = oacc[nt][r] / l_[r];
            size_t row = rowQ + hl * 8 + r;
            Ob[row * 1024 + h * DKH + nt * 16 + l16] = f2bf(v);
        }
    }
}

// ---------------------------------------------------------------------------
// Host-side orchestration
// ---------------------------------------------------------------------------
extern "C" void kernel_launch(void* const* d_in, const int* in_sizes, int n_in,
                              void* d_out, int out_size, void* d_ws, size_t ws_size,
                              hipStream_t stream) {
    const float* x_q = (const float*)d_in[0];
    const float* x_k = (const float*)d_in[1];
    const float* x_v = (const float*)d_in[2];
    const int*   msk = (const int*)d_in[3];
    const float* W_q = (const float*)d_in[4];
    const float* b_q = (const float*)d_in[5];
    const float* W_k = (const float*)d_in[6];
    const float* b_k = (const float*)d_in[7];
    const float* W_v = (const float*)d_in[8];
    const float* b_v = (const float*)d_in[9];
    const float* W_o = (const float*)d_in[10];
    const float* b_o = (const float*)d_in[11];

    const size_t NT = 4096;   // B*S
    const size_t D  = 1024;

    unsigned short* xq_b = (unsigned short*)d_ws;
    unsigned short* xk_b = xq_b + NT * D;
    unsigned short* xv_b = xk_b + NT * D;
    unsigned short* Wq_b = xv_b + NT * D;
    unsigned short* Wk_b = Wq_b + D * D;
    unsigned short* Wv_b = Wk_b + D * D;
    unsigned short* Wo_b = Wv_b + D * D;
    unsigned short* Qb   = Wo_b + D * D;
    unsigned short* Kb   = Qb + NT * D;
    unsigned short* Vb   = Kb + NT * D;
    unsigned short* Ob   = Vb + NT * D;

    const int nx = (int)(NT * D);   // 4194304
    const int nw = (int)(D * D);    // 1048576
    cvt_f32_bf16<<<nx / 2048, 256, 0, stream>>>(x_q, xq_b, nx);
    cvt_f32_bf16<<<nx / 2048, 256, 0, stream>>>(x_k, xk_b, nx);
    cvt_f32_bf16<<<nx / 2048, 256, 0, stream>>>(x_v, xv_b, nx);
    cvt_f32_bf16<<<nw / 2048, 256, 0, stream>>>(W_q, Wq_b, nw);
    cvt_f32_bf16<<<nw / 2048, 256, 0, stream>>>(W_k, Wk_b, nw);
    cvt_f32_bf16<<<nw / 2048, 256, 0, stream>>>(W_v, Wv_b, nw);
    cvt_f32_bf16<<<nw / 2048, 256, 0, stream>>>(W_o, Wo_b, nw);

    dim3 gg((unsigned)(D / 128), (unsigned)(NT / 128));   // (8, 32)
    gemm_bias<<<gg, 256, 0, stream>>>(xq_b, Wq_b, b_q, Qb, (int)NT, (int)D, (int)D);
    gemm_bias<<<gg, 256, 0, stream>>>(xk_b, Wk_b, b_k, Kb, (int)NT, (int)D, (int)D);
    gemm_bias<<<gg, 256, 0, stream>>>(xv_b, Wv_b, b_v, Vb, (int)NT, (int)D, (int)D);

    attn_flash<<<512, 256, 0, stream>>>(Qb, Kb, Vb, msk, Ob);

    gemm_bias<<<gg, 256, 0, stream>>>(Ob, Wo_b, b_o, (float*)d_out, (int)NT, (int)D, (int)D);
}
